// ChannelwiseHausdorffDistanceLoss_44796508897916
// MI455X (gfx1250) — compile-verified, matrix operand order checked by reading, not run on hardware
//
#include <hip/hip_runtime.h>

// ---------------------------------------------------------------------------
// Channelwise symmetric Hausdorff distance, MI455X (gfx1250, wave32, WMMA).
//
// d2[n,m] = |x_n|^2 + |y_m|^2 - 2 * (x_n . y_m)
// Gram matrix via v_wmma_f32_16x16x32_bf16 with the fp32 -> bf16(hi)+bf16(lo)
// split (3 WMMAs per 32-K chunk, fp32 accumulation). min/max/sqrt commute, so
// all reductions run on squared distances; sqrt happens once per (b,c).
//
// Pipeline:
//   1) init:  +inf into row/col running-min arrays (u32-atomicMin == float min
//             because d2 >= 0).
//   2) prep:  one pass over X,Y: split fp32 into bf16 hi/lo planes in d_ws
//             (same bytes/elem as fp32 -> no extra HBM pressure) AND exact
//             fp32 row norms. Removes ALL per-chunk VALU conversion from the
//             GEMM inner loop (previously redone 8x per tile).
//   3) tile:  double-buffered async global->LDS staging with
//             global_load_async_to_lds_b128 + s_wait_asynccnt (ASYNCcnt),
//             ds_load_b128 fragment gathers, 24 bf16 WMMAs per wave per chunk.
//   4) finalize: max / sqrt / mean.
//
// Workspace: 4 bf16 planes (4 * 48 MB) + norms + mins  ~= 202 MB of d_ws
// (about the size of the inputs). Problem constants: B=8, C=3, N=M=D=1024.
// ---------------------------------------------------------------------------

typedef __attribute__((ext_vector_type(16))) __bf16 v16bf;
typedef __attribute__((ext_vector_type(8)))  __bf16 v8bf;
typedef __attribute__((ext_vector_type(4)))  __bf16 v4bf;
typedef __attribute__((ext_vector_type(8)))  float  v8f;

#define BC_TOTAL   24          // B*C
#define NPTS       1024        // points per set
#define DDIM       1024        // point dimensionality (GEMM K)
#define ROWS_TOT   (BC_TOTAL * NPTS)
#define TILE       128         // output tile edge per workgroup
#define KCHUNK     32          // K per WMMA (bf16)
#define LDS_ST     40          // padded LDS row stride (elems): 80B = 20 banks,
                               // conflict-free for 16 consecutive rows x b128
#define PLANE_B    ((size_t)TILE * LDS_ST * 2)   // one bf16 plane tile in LDS
#define BUF_B      (4 * PLANE_B)                 // 4 planes per buffer (40960B)

// ---- CDNA5 async global->LDS copy (ASYNCcnt-tracked), via inline asm ------
__device__ __forceinline__ void async_ld_b128(unsigned lds_off,
                                              unsigned long long gaddr) {
    asm volatile("global_load_async_to_lds_b128 %0, %1, off"
                 :: "v"(lds_off), "v"(gaddr) : "memory");
}
__device__ __forceinline__ void wait_async0() {
    asm volatile("s_wait_asynccnt 0x0" ::: "memory");
}

// Gather one 16x32 bf16 WMMA fragment for this lane from an LDS tile stored
// row-major [row][k]. Per ISA 7.12.2 (16-bit A 16x32): lanes 0-15 hold
// K={0..7,16..23}, lanes 16-31 hold K={8..15,24..31} for row (lane&15).
__device__ __forceinline__ v16bf frag_load(const __bf16* rowbase, int kbase) {
    v8bf c0 = *reinterpret_cast<const v8bf*>(rowbase + kbase);
    v8bf c1 = *reinterpret_cast<const v8bf*>(rowbase + kbase + 16);
    return __builtin_shufflevector(c0, c1, 0,1,2,3,4,5,6,7,8,9,10,11,12,13,14,15);
}

// ---------------------------------------------------------------------------
// Kernel 1: +inf bits into running-min arrays.
// ---------------------------------------------------------------------------
__global__ void haus_init_kernel(unsigned* __restrict__ p) {
    p[blockIdx.x * blockDim.x + threadIdx.x] = 0x7F800000u; // +inf
}

// ---------------------------------------------------------------------------
// Kernel 2: prep. One wave per source row: split fp32 -> bf16 hi/lo planes
// and accumulate the exact fp32 row norm in the same pass.
// ---------------------------------------------------------------------------
__global__ __launch_bounds__(256) void haus_prep_kernel(
    const float* __restrict__ X, const float* __restrict__ Y,
    __bf16* __restrict__ Xhi, __bf16* __restrict__ Xlo,
    __bf16* __restrict__ Yhi, __bf16* __restrict__ Ylo,
    float* __restrict__ x2g, float* __restrict__ y2g) {
    const int lane = threadIdx.x & 31;
    const int wv   = threadIdx.x >> 5;
    const long row = (long)blockIdx.x * 8 + wv;        // 0 .. 2*ROWS_TOT-1
    const bool isX = row < ROWS_TOT;
    const long r   = isX ? row : row - ROWS_TOT;
    const float*  src = isX ? X : Y;
    __bf16* hip_ = isX ? Xhi : Yhi;
    __bf16* lop_ = isX ? Xlo : Ylo;
    float*  nrm  = isX ? x2g : y2g;

    const float4* p = reinterpret_cast<const float4*>(src + r * DDIM);
    float s = 0.f;
#pragma unroll
    for (int j = 0; j < 8; ++j) {                      // 8 * 32 lanes * 4 = 1024
        const int e4 = lane + 32 * j;                  // float4 index in row
        float4 v = p[e4];
        __bf16 hx = (__bf16)v.x, hy = (__bf16)v.y,
               hz = (__bf16)v.z, hw = (__bf16)v.w;
        v4bf h = {hx, hy, hz, hw};
        v4bf l = {(__bf16)(v.x - (float)hx), (__bf16)(v.y - (float)hy),
                  (__bf16)(v.z - (float)hz), (__bf16)(v.w - (float)hw)};
        *reinterpret_cast<v4bf*>(hip_ + r * DDIM + e4 * 4) = h;
        *reinterpret_cast<v4bf*>(lop_ + r * DDIM + e4 * 4) = l;
        s = fmaf(v.x, v.x, s); s = fmaf(v.y, v.y, s);
        s = fmaf(v.z, v.z, s); s = fmaf(v.w, v.w, s);
    }
    s += __shfl_xor(s, 16); s += __shfl_xor(s, 8);
    s += __shfl_xor(s, 4);  s += __shfl_xor(s, 2); s += __shfl_xor(s, 1);
    if (lane == 0) nrm[r] = s;
}

// ---------------------------------------------------------------------------
// Kernel 3: main tile kernel. Grid = 24 bc * 8 * 8 tiles. Block = 256 threads
// (8 waves); each wave owns a 32x64 sub-tile = 2x4 WMMA accumulators.
// Double-buffered async staging: per chunk each wave issues 8
// global_load_async_to_lds_b128 (16B/lane) covering its share of the 4 planes.
// ---------------------------------------------------------------------------
__global__ __launch_bounds__(256) void haus_tile_kernel(
    const __bf16* __restrict__ Xhi, const __bf16* __restrict__ Xlo,
    const __bf16* __restrict__ Yhi, const __bf16* __restrict__ Ylo,
    const float* __restrict__ x2g, const float* __restrict__ y2g,
    unsigned* __restrict__ rowmin, unsigned* __restrict__ colmin) {
    __shared__ __bf16 sbuf[2][4][TILE * LDS_ST];   // [buf][plane][row*ST+k]
    __shared__ float  sx2[TILE];
    __shared__ float  sy2[TILE];

    const int gx = blockIdx.x;
    const int tm = gx & 7;             // column tile (target rows)
    const int tn = (gx >> 3) & 7;      // row tile (input rows)
    const int bc = gx >> 6;            // (batch,channel)

    const int t    = threadIdx.x;
    const int lane = t & 31;
    const int wv   = t >> 5;
    const int r0   = (wv & 3) * 32;    // wave's rows within tile
    const int c0   = (wv >> 2) * 64;   // wave's cols within tile

    const size_t baseN = (size_t)bc * NPTS + (size_t)tn * TILE;
    const size_t baseM = (size_t)bc * NPTS + (size_t)tm * TILE;

    // Stage the fp32 norms for this tile into LDS (one element per thread).
    if (t < TILE) sx2[t] = x2g[baseN + t];
    else          sy2[t - TILE] = y2g[baseM + (t - TILE)];

    // ---- async-copy cursors: thread -> (plane, 1/64th of an 8KB plane) ----
    const int pl  = t >> 6;            // 0:Xhi 1:Xlo 2:Yhi 3:Ylo
    const int tid = t & 63;
    const __bf16* plane = (pl == 0) ? Xhi : (pl == 1) ? Xlo
                        : (pl == 2) ? Yhi : Ylo;
    const size_t prow = (pl < 2) ? baseN : baseM;
    const unsigned long long gbase =
        (unsigned long long)(plane + prow * DDIM);
    const unsigned lbase =
        (unsigned)(unsigned long long)(&sbuf[0][pl][0]);

    // Issue this wave's 8 async b128 loads for K-chunk kc into buffer b.
    auto issue_chunk = [&](int b, int kc) {
#pragma unroll
        for (int i = 0; i < 8; ++i) {
            const int g   = tid + 64 * i;    // 16B segment id (0..511)
            const int row = g >> 2;          // 0..127
            const int seg = g & 3;           // 16B segment within row chunk
            const unsigned long long ga =
                gbase + ((unsigned long long)row * DDIM + kc + seg * 8) * 2;
            const unsigned la =
                lbase + (unsigned)b * (unsigned)BUF_B
                      + (unsigned)(row * LDS_ST + seg * 8) * 2;
            async_ld_b128(la, ga);
        }
    };

    v8f acc[2][4];
#pragma unroll
    for (int i = 0; i < 2; ++i)
#pragma unroll
        for (int j = 0; j < 4; ++j)
            acc[i][j] = (v8f){0.f,0.f,0.f,0.f,0.f,0.f,0.f,0.f};

    issue_chunk(0, 0);                 // prime buffer 0

    for (int kc = 0; kc < DDIM; kc += KCHUNK) {
        const int cur = (kc >> 5) & 1;
        wait_async0();                 // this wave's LDS writes retired
        __syncthreads();               // all waves: cur readable, other buffer
                                       // finished being read last iteration
        if (kc + KCHUNK < DDIM) issue_chunk(cur ^ 1, kc + KCHUNK);

        const int kb = (lane < 16) ? 0 : 8;
        v16bf Ah[2], Al[2], Bh[4], Bl[4];
#pragma unroll
        for (int i = 0; i < 2; ++i) {
            const int m = r0 + 16 * i + (lane & 15);
            Ah[i] = frag_load(&sbuf[cur][0][m * LDS_ST], kb);
            Al[i] = frag_load(&sbuf[cur][1][m * LDS_ST], kb);
        }
#pragma unroll
        for (int j = 0; j < 4; ++j) {
            const int n = c0 + 16 * j + (lane & 15);
            Bh[j] = frag_load(&sbuf[cur][2][n * LDS_ST], kb);
            Bl[j] = frag_load(&sbuf[cur][3][n * LDS_ST], kb);
        }

        // x.y ~= hi.hi + hi.lo + lo.hi  (lo.lo term ~2^-16, dropped)
#pragma unroll
        for (int i = 0; i < 2; ++i)
#pragma unroll
            for (int j = 0; j < 4; ++j) {
                acc[i][j] = __builtin_amdgcn_wmma_f32_16x16x32_bf16(
                    false, Ah[i], false, Bh[j], (short)0, acc[i][j], false, false);
                acc[i][j] = __builtin_amdgcn_wmma_f32_16x16x32_bf16(
                    false, Ah[i], false, Bl[j], (short)0, acc[i][j], false, false);
                acc[i][j] = __builtin_amdgcn_wmma_f32_16x16x32_bf16(
                    false, Al[i], false, Bh[j], (short)0, acc[i][j], false, false);
            }
    }

    // ---- epilogue: acc -> clamped squared distances (ISA C-layout:
    //      VGPR r, lane l: row = r + 8*(l>=16), col = l&15) ----
    const int hs = (lane >> 4) * 8;
#pragma unroll
    for (int i = 0; i < 2; ++i)
#pragma unroll
        for (int j = 0; j < 4; ++j)
#pragma unroll
            for (int r = 0; r < 8; ++r) {
                const float nx2 = sx2[r0 + 16 * i + r + hs];
                const float ny2 = sy2[c0 + 16 * j + (lane & 15)];
                const float d2  = fmaf(-2.f, acc[i][j][r], nx2 + ny2);
                acc[i][j][r] = fmaxf(d2, 0.f);
            }

    // Row mins (min over this wave's 64 columns), then global atomic min.
#pragma unroll
    for (int i = 0; i < 2; ++i)
#pragma unroll
        for (int r = 0; r < 8; ++r) {
            float v = fminf(fminf(acc[i][0][r], acc[i][1][r]),
                            fminf(acc[i][2][r], acc[i][3][r]));
            v = fminf(v, __shfl_xor(v, 1));
            v = fminf(v, __shfl_xor(v, 2));
            v = fminf(v, __shfl_xor(v, 4));
            v = fminf(v, __shfl_xor(v, 8));
            if ((lane & 15) == r)
                atomicMin(&rowmin[baseN + r0 + 16 * i + r + hs],
                          __float_as_uint(v));
        }

    // Column mins (min over this wave's 32 rows), then global atomic min.
#pragma unroll
    for (int j = 0; j < 4; ++j) {
        float v = acc[0][j][0];
#pragma unroll
        for (int i = 0; i < 2; ++i)
#pragma unroll
            for (int r = 0; r < 8; ++r)
                v = fminf(v, acc[i][j][r]);
        v = fminf(v, __shfl_xor(v, 16));
        if (lane < 16)
            atomicMin(&colmin[baseM + c0 + 16 * j + lane], __float_as_uint(v));
    }
}

// ---------------------------------------------------------------------------
// Kernel 4: finalize. One block; wave w handles (b,c)=w: max over row/col
// mins, sqrt once, then mean over the 24 channels.
// ---------------------------------------------------------------------------
__global__ __launch_bounds__(1024) void haus_finalize_kernel(
    const unsigned* __restrict__ rowmin, const unsigned* __restrict__ colmin,
    float* __restrict__ out) {
    __shared__ float sh[32];
    const int lane = threadIdx.x & 31;
    const int wv   = threadIdx.x >> 5;
    float h = 0.f;
    if (wv < BC_TOTAL) {
        const float* rm = reinterpret_cast<const float*>(rowmin) + wv * NPTS;
        const float* cm = reinterpret_cast<const float*>(colmin) + wv * NPTS;
        float m = 0.f;
        for (int j = 0; j < NPTS / 32; ++j) {
            m = fmaxf(m, rm[lane + 32 * j]);
            m = fmaxf(m, cm[lane + 32 * j]);
        }
        m = fmaxf(m, __shfl_xor(m, 16)); m = fmaxf(m, __shfl_xor(m, 8));
        m = fmaxf(m, __shfl_xor(m, 4));  m = fmaxf(m, __shfl_xor(m, 2));
        m = fmaxf(m, __shfl_xor(m, 1));
        h = sqrtf(m);
    }
    if (lane == 0) sh[wv] = h;
    __syncthreads();
    if (threadIdx.x == 0) {
        float s = 0.f;
        for (int i = 0; i < BC_TOTAL; ++i) s += sh[i];
        out[0] = s / (float)BC_TOTAL;
    }
}

// ---------------------------------------------------------------------------
extern "C" void kernel_launch(void* const* d_in, const int* in_sizes, int n_in,
                              void* d_out, int out_size, void* d_ws, size_t ws_size,
                              hipStream_t stream) {
    (void)in_sizes; (void)n_in; (void)out_size; (void)ws_size;
    const float* X = (const float*)d_in[0];   // input  [8,3,1024,1024] f32
    const float* Y = (const float*)d_in[1];   // target [8,3,1024,1024] f32

    const size_t PLANE = (size_t)ROWS_TOT * DDIM;   // 25.2M elems
    float*    x2g    = (float*)d_ws;                         //  96 KB
    float*    y2g    = x2g + ROWS_TOT;                       //  96 KB
    unsigned* rowmin = (unsigned*)(y2g + ROWS_TOT);          //  96 KB
    unsigned* colmin = rowmin + ROWS_TOT;                    //  96 KB
    __bf16*   Xhi    = (__bf16*)(colmin + ROWS_TOT);         //  48 MB
    __bf16*   Xlo    = Xhi + PLANE;                          //  48 MB
    __bf16*   Yhi    = Xlo + PLANE;                          //  48 MB
    __bf16*   Ylo    = Yhi + PLANE;                          //  48 MB (~202 MB)

    // 1) +inf into rowmin+colmin (contiguous 2*ROWS_TOT entries).
    haus_init_kernel<<<(2 * ROWS_TOT) / 256, 256, 0, stream>>>(rowmin);
    // 2) bf16 hi/lo split planes + exact fp32 row norms, one wave per row.
    haus_prep_kernel<<<(2 * ROWS_TOT) / 8, 256, 0, stream>>>(
        X, Y, Xhi, Xlo, Yhi, Ylo, x2g, y2g);
    // 3) WMMA tile kernel: 24 bc * 8x8 tiles of 128x128, async-staged LDS.
    haus_tile_kernel<<<BC_TOTAL * 64, 256, 0, stream>>>(
        Xhi, Xlo, Yhi, Ylo, x2g, y2g, rowmin, colmin);
    // 4) max / sqrt / mean.
    haus_finalize_kernel<<<1, 1024, 0, stream>>>(rowmin, colmin, (float*)d_out);
}
